// CatanGraphEncoder_62792421868259
// MI455X (gfx1250) — compile-verified, matrix operand order checked by reading, not run on hardware
//
#include <hip/hip_runtime.h>
#include <cstddef>
#include <cstdint>

// ---------------------------------------------------------------------------
// CDNA5 (gfx1250) fused GCN encoder.
//  - All GEMMs via v_wmma_f32_16x16x32_bf16 (wave32, fp32 accumulate).
//  - Per-graph fused layer: aggregation expressed as dense 64x64 WMMA.
//  - Inter-layer activations kept in bf16 (memory-bound workload, 23.3 TB/s).
// ---------------------------------------------------------------------------

typedef __bf16 bf16_t;
typedef __attribute__((ext_vector_type(16))) __bf16 v16bf;
typedef __attribute__((ext_vector_type(8)))  float  v8f;

static constexpr int kB    = 2048;          // graphs
static constexpr int kNPG  = 54;            // nodes/graph (padded to 64)
static constexpr int kEPG  = 144;           // directed edges/graph
static constexpr int kN    = kB * kNPG;     // 110592 nodes
static constexpr int kNE   = kB * kEPG;     // 294912 edges
static constexpr int kIN   = 32;
static constexpr int kH    = 512;
static constexpr int kG    = 64;
static constexpr int kA    = 290;
static constexpr int kAPad = 384;           // 290 padded to 3*128 col tiles

__device__ __forceinline__ float bf2f(bf16_t x) {
    unsigned short u = __builtin_bit_cast(unsigned short, x);
    unsigned v = ((unsigned)u) << 16;
    return __builtin_bit_cast(float, v);
}
__device__ __forceinline__ bf16_t f2bf(float f) {
    unsigned u = __builtin_bit_cast(unsigned, f);
    unsigned r = (u + 0x7FFFu + ((u >> 16) & 1u)) >> 16;   // round-to-nearest-even
    return __builtin_bit_cast(bf16_t, (unsigned short)r);
}

union FragU { v16bf v; uint4 q[2]; };

// Load one 16x32 bf16 WMMA operand fragment.  The source matrix is stored
// row-major with the operand's "outer" index as rows: A as [M][K], B as [N][K].
// Wave32 layout (ISA 7.12.2): lane = 16*h + m holds row m; VGPR halves cover
// K = {h*8..h*8+7, 16+h*8..16+h*8+7}  -> two contiguous 16B chunks per lane.
__device__ __forceinline__ v16bf load_frag(const bf16_t* base, int ld,
                                           int row0, int k0, int lm, int lh) {
    const char* p = (const char*)(base + (size_t)(row0 + lm) * ld + k0 + lh * 8);
    FragU f;
    f.q[0] = *(const uint4*)(p);
    f.q[1] = *(const uint4*)(p + 32);
    return f.v;
}

__device__ __forceinline__ v8f zero8() {
    v8f v;
#pragma unroll
    for (int i = 0; i < 8; ++i) v[i] = 0.f;
    return v;
}

__device__ __forceinline__ v8f wmma_bf16(v16bf a, v16bf b, v8f c) {
    return __builtin_amdgcn_wmma_f32_16x16x32_bf16(false, a, false, b,
                                                   (short)0, c, false, false);
}

// ---------------------------------------------------------------------------
// Degree / symmetric normalization
// ---------------------------------------------------------------------------
__global__ void deg_init_k(int* deg) {
    int i = blockIdx.x * blockDim.x + threadIdx.x;
    if (i < kN) deg[i] = 1;                       // self loop
}
__global__ void deg_count_k(const int* __restrict__ edst, int* deg) {
    int i = blockIdx.x * blockDim.x + threadIdx.x;
    if (i < kNE) atomicAdd(&deg[edst[i]], 1);
}
__global__ void dinv_k(const int* __restrict__ deg, float* __restrict__ dinv) {
    int i = blockIdx.x * blockDim.x + threadIdx.x;
    if (i < kN) dinv[i] = rsqrtf((float)deg[i]);  // deg >= 1 always
}

// ---------------------------------------------------------------------------
// Weight prep: fp32 [K][Nsrc] -> bf16 transposed [Npad][K] (zero padded)
// ---------------------------------------------------------------------------
__global__ void transpose_bf16_k(const float* __restrict__ src, bf16_t* __restrict__ dst,
                                 int K, int Nsrc, int Npad) {
    int idx = blockIdx.x * blockDim.x + threadIdx.x;
    if (idx >= K * Npad) return;
    int n = idx / K, k = idx % K;
    float v = (n < Nsrc) ? src[(size_t)k * Nsrc + n] : 0.f;
    dst[(size_t)n * K + k] = f2bf(v);
}

// ---------------------------------------------------------------------------
// Fused GCN layer: one workgroup (8 wave32) per graph.
//   Xhat = A_norm(64x64) @ X(64xKIN)   (chunked over KIN, CH columns at a time)
//   H    = Xhat(64xKIN) @ W(KINx512)   (Wt pre-transposed bf16, L2 resident)
//   out  = relu(LN(H + bias))          (bf16)
// LDS: A32 16K | Abf 8K | dinv 256B | row stats 1K | Xt/Xh 2*CH*128B
// Note on determinism: LDS float atomics into A32 are order-invariant because
// every colliding add to cell (i,j) contributes the identical addend
// dinv[i]*dinv[j] -- repeated addition of the same value commutes exactly.
// ---------------------------------------------------------------------------
template <int KIN, bool INF32>
__global__ __launch_bounds__(256) void gcn_layer_k(
    const void* __restrict__ xin_raw,
    const int*  __restrict__ esrc, const int* __restrict__ edst,
    const float* __restrict__ dinv,
    const bf16_t* __restrict__ Wt,         // [512][KIN] bf16
    const float* __restrict__ bias,
    const float* __restrict__ gamma, const float* __restrict__ beta,
    bf16_t* __restrict__ xout)             // [kN][512] bf16
{
    constexpr int CH = (KIN < 128) ? KIN : 128;
    extern __shared__ __align__(16) char smem[];
    float*  A32  = (float*)(smem);                // 64*64 f32 staging
    bf16_t* Abf  = (bf16_t*)(smem + 16384);       // 64*64 bf16
    float*  dloc = (float*)(smem + 24576);        // 64
    float*  rsum = (float*)(smem + 24832);        // 64 rows x 2 col-groups
    float*  rsq  = (float*)(smem + 25344);
    bf16_t* Xt   = (bf16_t*)(smem + 25856);       // [CH][64]  (X^T chunk)
    bf16_t* Xh   = Xt + CH * 64;                  // [64][CH]  (Xhat chunk)

    const int g    = blockIdx.x;
    const int tid  = threadIdx.x;
    const int wave = tid >> 5;
    const int lane = tid & 31;
    const int lm   = lane & 15;
    const int lh   = lane >> 4;
    const int mt   = wave & 3;        // 16-row block
    const int grp  = wave >> 2;       // 256-col group

    for (int i = tid; i < 4096; i += 256) A32[i] = 0.f;
    if (tid < 64) dloc[tid] = (tid < kNPG) ? dinv[g * kNPG + tid] : 0.f;
    __syncthreads();

    // Dense A_norm scatter (order-invariant float atomics, see note above).
    for (int e = tid; e < kEPG; e += 256) {
        int s = esrc[g * kEPG + e] - g * kNPG;
        int d = edst[g * kEPG + e] - g * kNPG;
        atomicAdd(&A32[d * 64 + s], dloc[s] * dloc[d]);
    }
    if (tid < kNPG) atomicAdd(&A32[tid * 64 + tid], dloc[tid] * dloc[tid]);
    __syncthreads();
    for (int i = tid; i < 4096; i += 256) Abf[i] = f2bf(A32[i]);

    v8f acc[16];
#pragma unroll
    for (int t = 0; t < 16; ++t) acc[t] = zero8();

    const float*  xf = (const float*)xin_raw;
    const bf16_t* xb = (const bf16_t*)xin_raw;

    for (int kc = 0; kc < KIN; kc += CH) {
        __syncthreads();                           // Abf ready / Xh consumed
        // Prefetch next W K-chunk for this wave's column group into cache.
        if (kc + CH < KIN) {
            const bf16_t* pf = Wt + (size_t)(grp * 256 + lane * 8) * KIN + kc + CH;
            __builtin_prefetch(pf, 0, 1);
        }
        // Stage X^T chunk: Xt[c][node], zero-padded rows >= 54.
        for (int idx = tid; idx < CH * 64; idx += 256) {
            int c = idx % CH, node = idx / CH;
            float v = 0.f;
            if (node < kNPG) {
                size_t gi = (size_t)(g * kNPG + node) * KIN + kc + c;
                v = INF32 ? xf[gi] : bf2f(xb[gi]);
            }
            Xt[c * 64 + node] = f2bf(v);
        }
        __syncthreads();

        // Step 2: Xhat_chunk(64xCH) = A(64x64) @ X_chunk.  B operand = Xt [N][K].
        for (int t = wave; t < 4 * (CH / 16); t += 8) {
            int m2 = t & 3, n2 = t >> 2;
            v8f c2 = zero8();
#pragma unroll
            for (int k0 = 0; k0 < 64; k0 += 32) {
                v16bf a = load_frag(Abf, 64, m2 * 16, k0, lm, lh);
                v16bf b = load_frag(Xt, 64, n2 * 16, k0, lm, lh);
                c2 = wmma_bf16(a, b, c2);
            }
#pragma unroll
            for (int v = 0; v < 8; ++v) {
                int row = m2 * 16 + v + 8 * lh;
                Xh[row * CH + n2 * 16 + lm] = f2bf(c2[v]);
            }
        }
        __syncthreads();

        // Step 3: H += Xhat_chunk @ W_chunk.  Wave owns rows mt*16..+15 and the
        // 16 col tiles of group grp.  B frags preloaded in groups of 4 so the
        // compiler can clause 8 global_load_b128 per wait.
#pragma unroll
        for (int k0 = 0; k0 < CH; k0 += 32) {
            v16bf a = load_frag(Xh, CH, mt * 16, k0, lm, lh);
#pragma unroll
            for (int jb = 0; jb < 4; ++jb) {
                v16bf b[4];
#pragma unroll
                for (int t = 0; t < 4; ++t) {
                    int nt = grp * 16 + jb * 4 + t;
                    b[t] = load_frag(Wt, KIN, nt * 16, kc + k0, lm, lh);
                }
#pragma unroll
                for (int t = 0; t < 4; ++t)
                    acc[jb * 4 + t] = wmma_bf16(a, b[t], acc[jb * 4 + t]);
            }
        }
    }

    // Epilogue: bias, LayerNorm over 512, ReLU, bf16 store.
    float colbias[16], colg[16], colb[16];
#pragma unroll
    for (int j = 0; j < 16; ++j) {
        int col = grp * 256 + j * 16 + lm;
        colbias[j] = bias[col];
        colg[j]    = gamma[col];
        colb[j]    = beta[col];
    }
#pragma unroll
    for (int j = 0; j < 16; ++j)
#pragma unroll
        for (int v = 0; v < 8; ++v) acc[j][v] = acc[j][v] + colbias[j];

    // Per-row partial sums (deterministic: one slot per (row, col-group)).
#pragma unroll
    for (int v = 0; v < 8; ++v) {
        float s = 0.f, q = 0.f;
#pragma unroll
        for (int j = 0; j < 16; ++j) { float x = acc[j][v]; s += x; q += x * x; }
#pragma unroll
        for (int m = 1; m <= 8; m <<= 1) {
            s += __shfl_xor(s, m, 32);
            q += __shfl_xor(q, m, 32);
        }
        if (lm == 0) {
            int row = mt * 16 + v + 8 * lh;
            rsum[row * 2 + grp] = s;
            rsq[row * 2 + grp]  = q;
        }
    }
    __syncthreads();

#pragma unroll
    for (int v = 0; v < 8; ++v) {
        int row = mt * 16 + v + 8 * lh;
        if (row < kNPG) {
            float mu  = (rsum[row * 2] + rsum[row * 2 + 1]) * (1.f / 512.f);
            float msq = (rsq[row * 2] + rsq[row * 2 + 1]) * (1.f / 512.f);
            float rstd = rsqrtf(msq - mu * mu + 1e-5f);
            size_t base = ((size_t)(g * kNPG) + row) * kH;
#pragma unroll
            for (int j = 0; j < 16; ++j) {
                int col = grp * 256 + j * 16 + lm;
                float y = (acc[j][v] - mu) * rstd * colg[j] + colb[j];
                y = fmaxf(y, 0.f);
                xout[base + col] = f2bf(y);
            }
        }
    }
}

// ---------------------------------------------------------------------------
// Mean-pool 54 nodes + concat global features -> combined bf16 [2048][576]
// ---------------------------------------------------------------------------
__global__ void pool_k(const bf16_t* __restrict__ x, const float* __restrict__ gf,
                       bf16_t* __restrict__ comb) {
    int idx = blockIdx.x * blockDim.x + threadIdx.x;
    if (idx >= kB * (kH + kG)) return;
    int g = idx / (kH + kG), c = idx % (kH + kG);
    float v;
    if (c < kH) {
        float s = 0.f;
        for (int i = 0; i < kNPG; ++i) s += bf2f(x[((size_t)g * kNPG + i) * kH + c]);
        v = s * (1.f / (float)kNPG);
    } else {
        v = gf[(size_t)g * kG + (c - kH)];
    }
    comb[(size_t)g * (kH + kG) + c] = f2bf(v);
}

// ---------------------------------------------------------------------------
// Head GEMM: out(M x outN) = act(A(MxK) @ B(KxN) + bias).  Bt is [Npad][K] bf16.
// Block tile 64x128 (grid.x = M/64, grid.y = Npad/128), 8 wave32, 4 tiles/wave.
// ---------------------------------------------------------------------------
__global__ __launch_bounds__(256) void gemm_head_k(
    const bf16_t* __restrict__ Amat, const bf16_t* __restrict__ Bt,
    const float* __restrict__ bias, int biasN, int K, int relu,
    bf16_t* __restrict__ outb, float* __restrict__ outf, int outN)
{
    const int tid  = threadIdx.x;
    const int wave = tid >> 5, lane = tid & 31;
    const int lm = lane & 15, lh = lane >> 4;
    const int mrow0 = blockIdx.x * 64 + (wave & 3) * 16;
    const int ntg   = blockIdx.y * 8 + (wave >> 2) * 4;

    v8f acc[4];
#pragma unroll
    for (int j = 0; j < 4; ++j) acc[j] = zero8();

    for (int k0 = 0; k0 < K; k0 += 32) {
        v16bf a = load_frag(Amat, K, mrow0, k0, lm, lh);
        v16bf b[4];
#pragma unroll
        for (int j = 0; j < 4; ++j)
            b[j] = load_frag(Bt, K, (ntg + j) * 16, k0, lm, lh);
#pragma unroll
        for (int j = 0; j < 4; ++j)
            acc[j] = wmma_bf16(a, b[j], acc[j]);
    }

#pragma unroll
    for (int j = 0; j < 4; ++j) {
        int col = (ntg + j) * 16 + lm;
        float bv = (col < biasN) ? bias[col] : 0.f;
#pragma unroll
        for (int v = 0; v < 8; ++v) {
            int row = mrow0 + v + 8 * lh;
            float y = acc[j][v] + bv;
            if (relu) y = fmaxf(y, 0.f);
            if (col < outN) {
                if (outb) outb[(size_t)row * outN + col] = f2bf(y);
                else      outf[(size_t)row * outN + col] = y;
            }
        }
    }
}

// ---------------------------------------------------------------------------
extern "C" void kernel_launch(void* const* d_in, const int* in_sizes, int n_in,
                              void* d_out, int out_size, void* d_ws, size_t ws_size,
                              hipStream_t stream) {
    (void)in_sizes; (void)n_in; (void)out_size; (void)ws_size;

    const float* x0   = (const float*)d_in[0];        // [N][32]
    const int*   eidx = (const int*)d_in[1];          // [2][B*E]
    const float* gf   = (const float*)d_in[2];        // [B][64]
    const float* W0   = (const float*)d_in[4];
    const float* b0   = (const float*)d_in[5];
    const float* W1   = (const float*)d_in[6];
    const float* b1   = (const float*)d_in[7];
    const float* W2   = (const float*)d_in[8];
    const float* b2   = (const float*)d_in[9];
    const float* g0   = (const float*)d_in[10];
    const float* be0  = (const float*)d_in[11];
    const float* g1   = (const float*)d_in[12];
    const float* be1  = (const float*)d_in[13];
    const float* g2   = (const float*)d_in[14];
    const float* be2  = (const float*)d_in[15];
    const float* hW0  = (const float*)d_in[16];
    const float* hb0  = (const float*)d_in[17];
    const float* hW1  = (const float*)d_in[18];
    const float* hb1  = (const float*)d_in[19];
    const float* hW2  = (const float*)d_in[20];
    const float* hb2  = (const float*)d_in[21];

    const int* esrc = eidx;
    const int* edst = eidx + kNE;

    char* ws = (char*)d_ws;
    size_t off = 0;
    auto carve = [&](size_t bytes) -> char* {
        char* p = ws + off;
        off = (off + bytes + 255) & ~(size_t)255;
        return p;
    };
    int*    deg   = (int*)carve((size_t)kN * 4);
    float*  dinvp = (float*)carve((size_t)kN * 4);
    bf16_t* W0t   = (bf16_t*)carve((size_t)kH * kIN * 2);
    bf16_t* W1t   = (bf16_t*)carve((size_t)kH * kH * 2);
    bf16_t* W2t   = (bf16_t*)carve((size_t)kH * kH * 2);
    bf16_t* hW0t  = (bf16_t*)carve((size_t)kH * (kH + kG) * 2);
    bf16_t* hW1t  = (bf16_t*)carve((size_t)kH * kH * 2);
    bf16_t* hW2t  = (bf16_t*)carve((size_t)kAPad * kH * 2);
    bf16_t* x1    = (bf16_t*)carve((size_t)kN * kH * 2);
    bf16_t* x2    = (bf16_t*)carve((size_t)kN * kH * 2);
    bf16_t* comb  = (bf16_t*)carve((size_t)kB * (kH + kG) * 2);
    bf16_t* h1    = (bf16_t*)carve((size_t)kB * kH * 2);
    bf16_t* h2    = (bf16_t*)carve((size_t)kB * kH * 2);

    // --- degree / D^-1/2 ---
    deg_init_k<<<(kN + 255) / 256, 256, 0, stream>>>(deg);
    deg_count_k<<<(kNE + 255) / 256, 256, 0, stream>>>(edst, deg);
    dinv_k<<<(kN + 255) / 256, 256, 0, stream>>>(deg, dinvp);

    // --- one-time weight transpose+convert (bf16, [N][K] for B operand) ---
    transpose_bf16_k<<<(kIN * kH + 255) / 256, 256, 0, stream>>>(W0, W0t, kIN, kH, kH);
    transpose_bf16_k<<<(kH * kH + 255) / 256, 256, 0, stream>>>(W1, W1t, kH, kH, kH);
    transpose_bf16_k<<<(kH * kH + 255) / 256, 256, 0, stream>>>(W2, W2t, kH, kH, kH);
    transpose_bf16_k<<<((kH + kG) * kH + 255) / 256, 256, 0, stream>>>(hW0, hW0t, kH + kG, kH, kH);
    transpose_bf16_k<<<(kH * kH + 255) / 256, 256, 0, stream>>>(hW1, hW1t, kH, kH, kH);
    transpose_bf16_k<<<(kH * kAPad + 255) / 256, 256, 0, stream>>>(hW2, hW2t, kH, kA, kAPad);

    // --- 3 fused GCN layers (one workgroup per graph) ---
    constexpr int lds32  = 25856 + 32 * 256;    // 34048 B
    constexpr int lds512 = 25856 + 128 * 256;   // 58624 B
    gcn_layer_k<kIN, true ><<<kB, 256, lds32,  stream>>>(x0, esrc, edst, dinvp, W0t, b0, g0, be0, x1);
    gcn_layer_k<kH,  false><<<kB, 256, lds512, stream>>>(x1, esrc, edst, dinvp, W1t, b1, g1, be1, x2);
    gcn_layer_k<kH,  false><<<kB, 256, lds512, stream>>>(x2, esrc, edst, dinvp, W2t, b2, g2, be2, x1);

    // --- pool + head MLP ---
    pool_k<<<(kB * (kH + kG) + 255) / 256, 256, 0, stream>>>(x1, gf, comb);

    dim3 gh0(kB / 64, kH / 128);
    gemm_head_k<<<gh0, 256, 0, stream>>>(comb, hW0t, hb0, kH, kH + kG, 1, h1, nullptr, kH);
    gemm_head_k<<<gh0, 256, 0, stream>>>(h1, hW1t, hb1, kH, kH, 1, h2, nullptr, kH);
    dim3 gh2(kB / 64, kAPad / 128);
    gemm_head_k<<<gh2, 256, 0, stream>>>(h2, hW2t, hb2, kA, kH, 0, nullptr, (float*)d_out, kA);
}
